// MultiHeadedAttention_3_type_18631568130079
// MI455X (gfx1250) — matrix operand order repaired and checked by reading, not compiled, not run on hardware
//
#include <hip/hip_runtime.h>
#include <hip/hip_bf16.h>

typedef __attribute__((ext_vector_type(16))) _Float16 v16h;
typedef __attribute__((ext_vector_type(8)))  float    v8f;

#define BB   8
#define NN   128
#define CC   256
#define HH   8
#define KTOT 512                 // 2*C concatenated reduction
#define MTOT (BB * NN * NN)      // 131072 rows
#define NTILES (MTOT / 16)       // 8192 wave-tiles

// ---------------------------------------------------------------------------
// Kernel 1: attn = sigmoid(Q*Wq^T + K*Wk^T + b), via v_wmma_f32_16x16x32_f16.
// One wave per 16-row tile. W staged in LDS as ready-made f16 B-fragments.
// ---------------------------------------------------------------------------
__global__ __launch_bounds__(256) void attn_gemm_kernel(
    const float* __restrict__ q, const float* __restrict__ kt,
    const float* __restrict__ W, const float* __restrict__ bias,
    float* __restrict__ attn)
{
    // B-fragment staging: 16 k-steps * 32 lanes * 16 halves = 16 KB
    __shared__ __align__(32) _Float16 ldsB[16 * 32 * 16];

    const int tid = threadIdx.x;

    // Build f16 B fragments for all 16 K-steps. Fragment (s, lane):
    //   col n = lane&15 (h if <8 else zero column)
    //   halves u=0..15 correspond to K = 32*s + ((lane>>4)?8:0) + u + (u>=8 ? 8 : 0)
    for (int f = tid; f < 16 * 32; f += 256) {
        const int s    = f >> 5;
        const int lane = f & 31;
        const int n    = lane & 15;
        const int kk   = s * 32 + ((lane >> 4) & 1) * 8;
        _Float16* dst  = &ldsB[(size_t)f * 16];
        if (n < HH) {
            const float* wrow = W + (size_t)n * KTOT + kk;
            #pragma unroll
            for (int u = 0; u < 16; ++u) dst[u] = (_Float16)wrow[u + (u >> 3) * 8];
        } else {
            #pragma unroll
            for (int u = 0; u < 16; ++u) dst[u] = (_Float16)0.0f;
        }
    }
    __syncthreads();

    const int lane  = tid & 31;
    const int tile  = blockIdx.x * 8 + (tid >> 5);   // 8 waves / block
    if (tile >= NTILES) return;

    const int row0  = tile * 16;
    const int arow  = row0 + (lane & 15);            // this lane's A row
    const int cbase = (lane >> 4) * 8;               // K sub-group base per A layout

    v8f acc = {};

    const float* srcs[2] = { q, kt };                // [Q | K] along K dimension
    #pragma unroll
    for (int half = 0; half < 2; ++half) {
        const float* src = srcs[half] + (size_t)arow * CC;
        #pragma unroll
        for (int s8 = 0; s8 < 8; ++s8) {
            const int s  = half * 8 + s8;
            const int c0 = s8 * 32 + cbase;

            // A: halves 0..7  = K c0..c0+7 ; halves 8..15 = K c0+16..c0+23
            const float4* p0 = (const float4*)(src + c0);
            const float4* p1 = (const float4*)(src + c0 + 16);
            float4 x0 = p0[0], x1 = p0[1];
            float4 x2 = p1[0], x3 = p1[1];

            v16h av;
            av[0]  = (_Float16)x0.x; av[1]  = (_Float16)x0.y;
            av[2]  = (_Float16)x0.z; av[3]  = (_Float16)x0.w;
            av[4]  = (_Float16)x1.x; av[5]  = (_Float16)x1.y;
            av[6]  = (_Float16)x1.z; av[7]  = (_Float16)x1.w;
            av[8]  = (_Float16)x2.x; av[9]  = (_Float16)x2.y;
            av[10] = (_Float16)x2.z; av[11] = (_Float16)x2.w;
            av[12] = (_Float16)x3.x; av[13] = (_Float16)x3.y;
            av[14] = (_Float16)x3.z; av[15] = (_Float16)x3.w;

            v16h bv = *(const v16h*)&ldsB[((size_t)s * 32 + lane) * 16];

            acc = __builtin_amdgcn_wmma_f32_16x16x32_f16(
                false, av, false, bv, (short)0, acc, false, false);
        }
    }

    // D layout: lane's column n = lane&15 (valid h if < 8),
    //           VGPR r holds row = row0 + r + (lane<16 ? 0 : 8)
    const int h = lane & 15;
    if (h < HH) {
        const float bb   = bias[h];
        const int   roff = (lane < 16) ? 0 : 8;
        #pragma unroll
        for (int r = 0; r < 8; ++r) {
            float x = acc[r] + bb;
            float sg = __fdividef(1.0f, 1.0f + __expf(-x));
            attn[(size_t)(row0 + roff + r) * HH + h] = sg;
        }
    }
}

// ---------------------------------------------------------------------------
// Kernel 2a: zero the pos mask (128 entries) each call
// ---------------------------------------------------------------------------
__global__ void zero_pos_kernel(float* __restrict__ pos)
{
    pos[threadIdx.x] = 0.0f;
}

// ---------------------------------------------------------------------------
// Kernel 2b: exact top-k membership via rank. j is in top-64 of slice v[:]
// iff #{j': v[j']>v[j]} + #{j'<j: v[j']==v[j]} < 64 (stable tie-break, as
// jax.lax.top_k). pos[j]=1 if any (b,i,h,mask) slice selects j.
// One block per (b,i); threads = j.
// ---------------------------------------------------------------------------
__global__ __launch_bounds__(128) void topk_rank_kernel(
    const float* __restrict__ attn, const float* __restrict__ m1,
    const float* __restrict__ m2, float* __restrict__ pos)
{
    const int bi = blockIdx.x;          // b*N + i
    const int j  = threadIdx.x;         // 0..127

    __shared__ float s1[NN];
    __shared__ float s2[NN];

    const float mm1 = m1[(size_t)bi * NN + j];
    const float mm2 = m2[(size_t)bi * NN + j];
    const float* arow = attn + ((size_t)bi * NN + j) * HH;

    bool hit = false;
    for (int h = 0; h < HH; ++h) {
        const float v  = arow[h];
        const float a1 = v * mm1;
        const float a2 = v * mm2;
        __syncthreads();                // protect LDS from previous iteration
        s1[j] = a1;
        s2[j] = a2;
        __syncthreads();
        int c1 = 0, c2 = 0;
        for (int jp = 0; jp < NN; ++jp) {
            const float w1 = s1[jp];
            const float w2 = s2[jp];
            c1 += (w1 > a1) || (w1 == a1 && jp < j);
            c2 += (w2 > a2) || (w2 == a2 && jp < j);
        }
        if (c1 < (NN / 2) || c2 < (NN / 2)) hit = true;
    }
    if (hit) pos[j] = 1.0f;             // benign same-value race across blocks
}

// ---------------------------------------------------------------------------
// Kernel 3: out[b,i,j,:] *= (m1[b,i,j]+m2[b,i,j]) * pos[j]   (in place)
// ---------------------------------------------------------------------------
__global__ __launch_bounds__(256) void finalize_kernel(
    float* __restrict__ out, const float* __restrict__ m1,
    const float* __restrict__ m2, const float* __restrict__ pos)
{
    const int bij = blockIdx.x * 256 + threadIdx.x;   // 0..MTOT-1
    const int j   = bij & (NN - 1);
    const float scale = (m1[bij] + m2[bij]) * pos[j];

    float4* o = (float4*)(out + (size_t)bij * HH);
    float4 v0 = o[0], v1 = o[1];
    v0.x *= scale; v0.y *= scale; v0.z *= scale; v0.w *= scale;
    v1.x *= scale; v1.y *= scale; v1.z *= scale; v1.w *= scale;
    o[0] = v0; o[1] = v1;
}

// ---------------------------------------------------------------------------
extern "C" void kernel_launch(void* const* d_in, const int* in_sizes, int n_in,
                              void* d_out, int out_size, void* d_ws, size_t ws_size,
                              hipStream_t stream)
{
    const float* q    = (const float*)d_in[0];   // (B,N,N,C)
    const float* kt   = (const float*)d_in[1];   // (B,N,N,C)
    const float* m1   = (const float*)d_in[2];   // (B,N,N)
    const float* m2   = (const float*)d_in[3];   // (B,N,N)
    const float* W    = (const float*)d_in[4];   // (H, 2C)
    const float* bias = (const float*)d_in[5];   // (H,)
    float* out = (float*)d_out;                  // (B,N,N,H) — also attn scratch
    float* pos = (float*)d_ws;                   // 128 floats

    // 1) attn = sigmoid(QWq + KWk + b) -> d_out
    attn_gemm_kernel<<<NTILES / 8, 256, 0, stream>>>(q, kt, W, bias, out);
    // 2) pos mask from exact top-64 ranks
    zero_pos_kernel<<<1, NN, 0, stream>>>(pos);
    topk_rank_kernel<<<BB * NN, NN, 0, stream>>>(out, m1, m2, pos);
    // 3) out = attn * (m1+m2) * pos  (in place)
    finalize_kernel<<<MTOT / 256, 256, 0, stream>>>(out, m1, m2, pos);
}
